// net_82901458747706
// MI455X (gfx1250) — matrix operand order repaired and checked by reading, not compiled, not run on hardware
//
#include <hip/hip_runtime.h>
#include <math.h>

// ---------------- constants ----------------
#define TT   513          // T_FRAMES
#define CC   514          // C = 2*FREQ
#define TP   544          // pad32(513)
#define CP   544          // pad32(514)
#define FP   2080         // pad32(2052 / 2056)
#define F4T  2052
#define F4C  2056
#define NFFT 512
#define HOP  128
#define FRQ  257
#define SEQN 65536
#define BB   16
#define NFR  (BB*TT)      // 8208 frames
#define NFRP 8224         // pad32(8208) rows for WMMA A-operands
#define LBLK 6
#define EPSR 1e-6f

#define EPI_SILU  1
#define EPI_BF16  2
#define EPI_TRANS 4
#define EPI_ACC   8

typedef __bf16 bf16;
typedef __attribute__((ext_vector_type(16))) __bf16 v16bf;
typedef __attribute__((ext_vector_type(8)))  float  v8f;

__device__ __forceinline__ bf16 f2bf(float f) {
  unsigned u = __builtin_bit_cast(unsigned, f);
  u += 0x7fffu + ((u >> 16) & 1u);               // round-to-nearest-even
  unsigned short h = (unsigned short)(u >> 16);
  return __builtin_bit_cast(bf16, h);
}
__device__ __forceinline__ float siluf(float x) { return x / (1.f + __expf(-x)); }
__device__ __forceinline__ float winf(int j) {
  return 0.5f - 0.5f * __cosf((float)j * (6.283185307179586f / (float)NFFT));
}

template<int BS>
__device__ float redsum(float v) {
  __shared__ float sm[BS];
  int t = threadIdx.x;
  sm[t] = v; __syncthreads();
  for (int s = BS >> 1; s > 0; s >>= 1) { if (t < s) sm[t] += sm[t + s]; __syncthreads(); }
  float r = sm[0]; __syncthreads(); return r;
}
template<int BS>
__device__ float redmax(float v) {
  __shared__ float sm[BS];
  int t = threadIdx.x;
  sm[t] = v; __syncthreads();
  for (int s = BS >> 1; s > 0; s >>= 1) { if (t < s) sm[t] = fmaxf(sm[t], sm[t + s]); __syncthreads(); }
  float r = sm[0]; __syncthreads(); return r;
}

// ---------------- WMMA fragment loads (ISA 7.12.2 layouts) ----------------
// A 16x32 bf16: lane half0 holds K 0-7 & 16-23, half1 holds K 8-15 & 24-31 (per M=lane%16 row)
__device__ __forceinline__ v16bf load_afrag(const bf16* rowp, int k, int hf) {
  union { v16bf v; unsigned u[8]; } f;
  const unsigned* p = (const unsigned*)(rowp + k + hf * 8);
  f.u[0] = p[0]; f.u[1] = p[1]; f.u[2] = p[2];  f.u[3] = p[3];
  f.u[4] = p[8]; f.u[5] = p[9]; f.u[6] = p[10]; f.u[7] = p[11];
  return f.v;
}
// B 32x16 bf16: lane half0 holds K 0-7 & 16-23, half1 holds K 8-15 & 24-31 (per N=lane%16 col)
__device__ __forceinline__ v16bf load_bfrag(const bf16* colp, int k, int hf) {
  union { v16bf v; unsigned u[8]; } f;
  const unsigned* p = (const unsigned*)(colp + k + hf * 8);
#pragma unroll
  for (int j = 0; j < 8; ++j) f.u[j] = p[j + (j >> 2) * 4];
  return f.v;
}

// ---------------- WMMA GEMM:  D[M,N] = act(alpha*A[M,K]*W[N,K]^T + bias) ----------------
// A, W: bf16 row-major, K padded to mult-of-32 (zero filled), rows padded to mult-of-32.
// One 32x32 output tile (2x2 WMMA register block) per wave32; 4 waves per block.
__global__ void __launch_bounds__(128)
gemm_bf16_wmma_kernel(const bf16* __restrict__ A, long sAb, int lda,
                      const bf16* __restrict__ W, long sWb, int ldw,
                      void* __restrict__ D, long sDb, int ldd,
                      const float* __restrict__ bias,
                      int M, int N, int K, float alpha, int flags)
{
  const int lane = threadIdx.x & 31;
  const int wave = threadIdx.x >> 5;
  const int tiles_n = (N + 31) >> 5;
  const int tiles_m = (M + 31) >> 5;
  int tile = blockIdx.x * 4 + wave;
  if (tile >= tiles_m * tiles_n) return;
  const int tm = tile / tiles_n;
  const int tn = tile - tm * tiles_n;
  const int b  = blockIdx.y;
  const int hf = lane >> 4;       // lane half
  const int lr = lane & 15;
  const bf16* arow0 = A + (long)b * sAb + (long)(tm * 32 + lr) * lda;
  const bf16* arow1 = arow0 + (long)16 * lda;
  const bf16* wcol0 = W + (long)b * sWb + (long)(tn * 32 + lr) * ldw;
  const bf16* wcol1 = wcol0 + (long)16 * ldw;
  v8f acc[2][2];
#pragma unroll
  for (int i = 0; i < 2; ++i)
#pragma unroll
    for (int j = 0; j < 2; ++j)
      acc[i][j] = (v8f){0.f,0.f,0.f,0.f,0.f,0.f,0.f,0.f};
  for (int k = 0; k < K; k += 32) {
    v16bf a0 = load_afrag(arow0, k, hf);
    v16bf a1 = load_afrag(arow1, k, hf);
    v16bf w0 = load_bfrag(wcol0, k, hf);
    v16bf w1 = load_bfrag(wcol1, k, hf);
    if (k + 32 < K) {                      // near-cache prefetch, one K-step ahead
      __builtin_prefetch(arow0 + k + 32, 0, 3);
      __builtin_prefetch(arow1 + k + 32, 0, 3);
      __builtin_prefetch(wcol0 + k + 32, 0, 3);
      __builtin_prefetch(wcol1 + k + 32, 0, 3);
    }
    acc[0][0] = __builtin_amdgcn_wmma_f32_16x16x32_bf16(false, a0, false, w0, (short)0, acc[0][0], false, false);
    acc[0][1] = __builtin_amdgcn_wmma_f32_16x16x32_bf16(false, a0, false, w1, (short)0, acc[0][1], false, false);
    acc[1][0] = __builtin_amdgcn_wmma_f32_16x16x32_bf16(false, a1, false, w0, (short)0, acc[1][0], false, false);
    acc[1][1] = __builtin_amdgcn_wmma_f32_16x16x32_bf16(false, a1, false, w1, (short)0, acc[1][1], false, false);
  }
#pragma unroll
  for (int ni = 0; ni < 2; ++ni) {
    const int col = tn * 32 + ni * 16 + lr;
    float bv = 0.f;
    if (bias != nullptr && col < N) bv = bias[col];
#pragma unroll
    for (int mi = 0; mi < 2; ++mi) {
#pragma unroll
      for (int r = 0; r < 8; ++r) {
        int row = tm * 32 + mi * 16 + hf * 8 + r;   // C/D layout: VGPR r -> M = r + half*8
        if (row < M && col < N) {
          float v = acc[mi][ni][r] * alpha + bv;
          if (flags & EPI_SILU) v = siluf(v);
          long offd = (flags & EPI_TRANS) ? ((long)col * ldd + row) : ((long)row * ldd + col);
          if (flags & EPI_BF16) {
            ((bf16*)D)[(long)b * sDb + offd] = f2bf(v);
          } else {
            float* p = (float*)D + (long)b * sDb + offd;
            if (flags & EPI_ACC) *p += v; else *p = v;
          }
        }
      }
    }
  }
}

// ---------------- helper / elementwise kernels ----------------
__global__ void pack_w_kernel(const float* __restrict__ src, int N, int K, int ld,
                              bf16* __restrict__ dst, int Np, int Kp)
{
  long id = (long)blockIdx.x * 256 + threadIdx.x;
  long tot = (long)Np * Kp;
  if (id >= tot) return;
  int n = (int)(id / Kp), k = (int)(id % Kp);
  float v = (n < N && k < K) ? src[(long)n * ld + k] : 0.f;
  dst[id] = f2bf(v);
}

__global__ void build_basis_fwd_kernel(bf16* __restrict__ b1)   // [CP][NFFT]
{
  long id = (long)blockIdx.x * 256 + threadIdx.x;
  if (id >= (long)CP * NFFT) return;
  int n = (int)(id >> 9), k = (int)(id & (NFFT - 1));
  float v = 0.f;
  if (n < FRQ)            { int r = (n * k) & (NFFT - 1); v =  __cosf((float)r * (6.283185307179586f / NFFT)); }
  else if (n < 2 * FRQ)   { int m = n - FRQ; int r = (m * k) & (NFFT - 1); v = -__sinf((float)r * (6.283185307179586f / NFFT)); }
  b1[id] = f2bf(v);
}

__global__ void build_basis_inv_kernel(bf16* __restrict__ b2)   // [NFFT][CP], window folded in
{
  long id = (long)blockIdx.x * 256 + threadIdx.x;
  if (id >= (long)NFFT * CP) return;
  int j = (int)(id / CP), n = (int)(id % CP);
  float w = winf(j) * (1.f / (float)NFFT);
  float v = 0.f;
  if (n < FRQ) {
    float cn = (n == 0 || n == FRQ - 1) ? 1.f : 2.f;
    int r = (n * j) & (NFFT - 1);
    v = cn * __cosf((float)r * (6.283185307179586f / NFFT)) * w;
  } else if (n < 2 * FRQ) {
    int m = n - FRQ;
    if (m != 0 && m != FRQ - 1) {
      int r = (m * j) & (NFFT - 1);
      v = -2.f * __sinf((float)r * (6.283185307179586f / NFFT)) * w;
    }
  }
  b2[id] = f2bf(v);
}

__global__ void build_frames_kernel(const float* __restrict__ x, bf16* __restrict__ frm)
{
  long id = (long)blockIdx.x * 256 + threadIdx.x;
  if (id >= (long)NFRP * NFFT) return;
  int f = (int)(id >> 9), j = (int)(id & (NFFT - 1));
  if (f >= NFR) { frm[id] = f2bf(0.f); return; }    // 32-row pad for WMMA A-operand
  int b = f / TT, t = f % TT;
  int idx = t * HOP + j - NFFT / 2;                 // reflect pad
  if (idx < 0) idx = -idx;
  if (idx >= SEQN) idx = 2 * SEQN - 2 - idx;
  frm[id] = f2bf(x[(long)b * SEQN + idx] * winf(j));
}

__global__ void magphase_kernel(const float* __restrict__ z, float* __restrict__ feat)
{
  long id = (long)blockIdx.x * 256 + threadIdx.x;
  if (id >= (long)NFR * FRQ) return;
  int f = (int)(id / FRQ), n = (int)(id % FRQ);
  float re = z[(long)f * CC + n];
  float im = z[(long)f * CC + FRQ + n];
  feat[(long)f * CC + n]       = sqrtf(re * re + im * im);
  feat[(long)f * CC + FRQ + n] = atan2f(im, re);
}

__global__ void rms_axis1_kernel(const float* __restrict__ x, const float* __restrict__ w,
                                 float* __restrict__ a)
{
  int c = blockIdx.x, b = blockIdx.y, tid = threadIdx.x;
  float s = 0.f;
  for (int t = tid; t < TT; t += 128) { float v = x[((long)b * TT + t) * CC + c]; s += v * v; }
  s = redsum<128>(s);
  float rs = rsqrtf(s * (1.f / TT) + EPSR);
  for (int t = tid; t < TT; t += 128)
    a[((long)b * TT + t) * CC + c] = x[((long)b * TT + t) * CC + c] * rs * w[t];
}

__global__ void decompose_pack_kernel(const float* __restrict__ a, bf16* __restrict__ trT,
                                      bf16* __restrict__ seT, bf16* __restrict__ aT)
{
  long id = (long)blockIdx.x * 256 + threadIdx.x;
  if (id >= (long)CP * TP) return;
  int cp = (int)(id / TP), tp = (int)(id % TP), b = blockIdx.y;
  float tr = 0.f, se = 0.f, av = 0.f;
  if (cp < CC && tp < TT) {
    av = a[((long)b * TT + tp) * CC + cp];
    for (int d = -2; d <= 2; ++d) {
      int t2 = tp + d; t2 = t2 < 0 ? 0 : (t2 >= TT ? TT - 1 : t2);
      tr += a[((long)b * TT + t2) * CC + cp];
    }
    tr *= 0.2f;
    se = av - tr;
  }
  long o = (long)b * CP * TP + id;
  trT[o] = f2bf(tr); seT[o] = f2bf(se); aT[o] = f2bf(av);
}

__global__ void lam_kernel(const float* __restrict__ q1, const float* __restrict__ k1,
                           const float* __restrict__ q2, const float* __restrict__ k2,
                           float lam_init, float* __restrict__ lamp)
{
  int tid = threadIdx.x;
  float s1 = 0.f, s2 = 0.f;
  for (int t = tid; t < TT; t += 512) { s1 += q1[t] * k1[t]; s2 += q2[t] * k2[t]; }
  s1 = redsum<512>(s1); s2 = redsum<512>(s2);
  if (tid == 0) lamp[0] = __expf(s1) - __expf(s2) + lam_init;
}

__global__ void softmax_combine_kernel(const float* __restrict__ a1, const float* __restrict__ a2,
                                       const float* __restrict__ lamp, bf16* __restrict__ dmat)
{
  int r = blockIdx.x, b = blockIdx.y, tid = threadIdx.x;
  bf16* drow = dmat + ((long)b * CP + r) * CP;
  if (r >= CC) { for (int c = tid; c < CP; c += 256) drow[c] = f2bf(0.f); return; }
  const float* r1 = a1 + ((long)b * CC + r) * CC;
  const float* r2 = a2 + ((long)b * CC + r) * CC;
  float m1 = -3.4e38f, m2 = -3.4e38f;
  for (int c = tid; c < CC; c += 256) { m1 = fmaxf(m1, r1[c]); m2 = fmaxf(m2, r2[c]); }
  m1 = redmax<256>(m1); m2 = redmax<256>(m2);
  float s1 = 0.f, s2 = 0.f;
  for (int c = tid; c < CC; c += 256) { s1 += __expf(r1[c] - m1); s2 += __expf(r2[c] - m2); }
  s1 = redsum<256>(s1); s2 = redsum<256>(s2);
  float lam = lamp[0];
  float i1 = 1.f / s1, i2 = lam / s2;
  for (int c = tid; c < CP; c += 256) {
    float v = 0.f;
    if (c < CC) v = __expf(r1[c] - m1) * i1 - __expf(r2[c] - m2) * i2;
    drow[c] = f2bf(v);
  }
}

__global__ void rms_add_tr_kernel(const float* __restrict__ attn, const float* __restrict__ nw,
                                  float oml, float* __restrict__ x)
{
  int c = blockIdx.x, b = blockIdx.y, tid = threadIdx.x;
  const float* ar = attn + ((long)b * CC + c) * TT;
  float s = 0.f;
  for (int t = tid; t < TT; t += 128) { float v = ar[t]; s += v * v; }
  s = redsum<128>(s);
  float rs = rsqrtf(s * (1.f / TT) + EPSR) * oml;
  for (int t = tid; t < TT; t += 128)
    x[((long)b * TT + t) * CC + c] += ar[t] * rs * nw[t];
}

__global__ void rms_t_pack_kernel(const float* __restrict__ x, const float* __restrict__ w,
                                  bf16* __restrict__ hA)
{
  int c = blockIdx.x, b = blockIdx.y, tid = threadIdx.x;
  bf16* row = hA + ((long)b * CP + c) * TP;
  if (c >= CC) { for (int t = tid; t < TP; t += 128) row[t] = f2bf(0.f); return; }
  float s = 0.f;
  for (int t = tid; t < TT; t += 128) { float v = x[((long)b * TT + t) * CC + c]; s += v * v; }
  s = redsum<128>(s);
  float rs = rsqrtf(s * (1.f / TT) + EPSR);
  for (int t = tid; t < TP; t += 128) {
    float v = (t < TT) ? x[((long)b * TT + t) * CC + c] * rs * w[t] : 0.f;
    row[t] = f2bf(v);
  }
}

__global__ void ln_t_pack_kernel(const float* __restrict__ x, const float* __restrict__ g,
                                 const float* __restrict__ bt, bf16* __restrict__ hA)
{
  int c = blockIdx.x, b = blockIdx.y, tid = threadIdx.x;
  bf16* row = hA + ((long)b * CP + c) * TP;
  if (c >= CC) { for (int t = tid; t < TP; t += 128) row[t] = f2bf(0.f); return; }
  float s = 0.f, s2 = 0.f;
  for (int t = tid; t < TT; t += 128) { float v = x[((long)b * TT + t) * CC + c]; s += v; s2 += v * v; }
  s = redsum<128>(s); s2 = redsum<128>(s2);
  float mu = s * (1.f / TT);
  float var = s2 * (1.f / TT) - mu * mu;
  float rs = rsqrtf(var + 1e-5f);
  for (int t = tid; t < TP; t += 128) {
    float v = 0.f;
    if (t < TT) v = (x[((long)b * TT + t) * CC + c] - mu) * rs * g[t] + bt[t];
    row[t] = f2bf(v);
  }
}

__global__ void polar_pack_kernel(const float* __restrict__ fin, bf16* __restrict__ rim)
{
  long id = (long)blockIdx.x * 256 + threadIdx.x;
  if (id >= (long)NFRP * CP) return;
  int f = (int)(id / CP), n = (int)(id % CP);
  float v = 0.f;
  if (f < NFR && n < 2 * FRQ) {
    int m = (n < FRQ) ? n : (n - FRQ);
    float s = fin[(long)f * CC + m];
    float p = fin[(long)f * CC + FRQ + m];
    v = (n < FRQ) ? s * __cosf(p) : s * __sinf(p);
  }
  rim[id] = f2bf(v);
}

__global__ void overlap_add_kernel(const float* __restrict__ ofr, float* __restrict__ out)
{
  long id = (long)blockIdx.x * 256 + threadIdx.x;
  if (id >= (long)BB * SEQN) return;
  int b = (int)(id / SEQN), s = (int)(id % SEQN);
  int i = s + NFFT / 2;
  int t1 = i >> 7; if (t1 > TT - 1) t1 = TT - 1;
  int lo = i - (NFFT - 1);
  int t0 = lo <= 0 ? 0 : ((lo + HOP - 1) >> 7);
  float y = 0.f, ws = 0.f;
  for (int t = t0; t <= t1; ++t) {
    int j = i - t * HOP;
    y += ofr[((long)b * TT + t) * NFFT + j];
    float w = winf(j);
    ws += w * w;
  }
  out[id] = y / fmaxf(ws, 1e-11f);
}

// ---------------- host helpers ----------------
static inline void launch_gemm(hipStream_t st,
                               const void* A, long sAb, int lda,
                               const void* W, long sWb, int ldw,
                               void* D, long sDb, int ldd,
                               const float* bias, int M, int N, int K,
                               float alpha, int flags, int nb)
{
  int tiles = ((M + 31) / 32) * ((N + 31) / 32);
  dim3 g((unsigned)((tiles + 3) / 4), (unsigned)nb);
  gemm_bf16_wmma_kernel<<<g, 128, 0, st>>>((const bf16*)A, sAb, lda,
                                           (const bf16*)W, sWb, ldw,
                                           D, sDb, ldd, bias, M, N, K, alpha, flags);
}

static inline void launch_packw(hipStream_t st, const float* src, int N, int K, int ld,
                                void* dst, int Np, int Kp)
{
  long tot = (long)Np * Kp;
  pack_w_kernel<<<(unsigned)((tot + 255) / 256), 256, 0, st>>>(src, N, K, ld, (bf16*)dst, Np, Kp);
}

extern "C" void kernel_launch(void* const* d_in, const int* in_sizes, int n_in,
                              void* d_out, int out_size, void* d_ws, size_t ws_size,
                              hipStream_t stream)
{
  (void)in_sizes; (void)n_in; (void)out_size; (void)ws_size;
  const float* IN_x   = (const float*)d_in[0];
  const float* W_lnat = (const float*)d_in[1];
  const float* W_q12w = (const float*)d_in[2];
  const float* W_q12b = (const float*)d_in[3];
  const float* W_k12w = (const float*)d_in[4];
  const float* W_k12b = (const float*)d_in[5];
  const float* W_vw   = (const float*)d_in[6];
  const float* W_vb   = (const float*)d_in[7];
  const float* W_lq1  = (const float*)d_in[8];
  const float* W_lk1  = (const float*)d_in[9];
  const float* W_lq2  = (const float*)d_in[10];
  const float* W_lk2  = (const float*)d_in[11];
  const float* W_nw   = (const float*)d_in[12];
  const float* W_lnff = (const float*)d_in[13];
  const float* W_ff1w = (const float*)d_in[14];
  const float* W_ff1b = (const float*)d_in[15];
  const float* W_ff2w = (const float*)d_in[16];
  const float* W_ff2b = (const float*)d_in[17];
  const float* W_lng  = (const float*)d_in[18];
  const float* W_lnb  = (const float*)d_in[19];
  const float* W_llw1 = (const float*)d_in[20];
  const float* W_llb1 = (const float*)d_in[21];
  const float* W_llw2 = (const float*)d_in[22];
  const float* W_llb2 = (const float*)d_in[23];
  const float* W_l2w1 = (const float*)d_in[24];
  const float* W_l2b1 = (const float*)d_in[25];
  const float* W_l2w2 = (const float*)d_in[26];
  const float* W_l2b2 = (const float*)d_in[27];

  // ---- workspace carve (256B aligned bump allocator) ----
  size_t off = 0;
  auto AL = [&](size_t nbytes) -> void* {
    off = (off + 255) & ~(size_t)255;
    void* r = (char*)d_ws + off;
    off += nbytes;
    return r;
  };
  const size_t SQ2 = (size_t)CP * TP;            // 544*544 elements
  float* f_x   = (float*)AL((size_t)BB * TT * CC * 4);
  float* f_a   = (float*)AL((size_t)BB * TT * CC * 4);
  float* f_a1  = (float*)AL((size_t)BB * CC * CC * 4);
  float* f_a2  = (float*)AL((size_t)BB * CC * CC * 4);
  float* f_at  = (float*)AL((size_t)BB * CC * TT * 4);
  float* f_z   = (float*)AL((size_t)NFR * CC * 4);
  float* f_ofr = (float*)AL((size_t)NFR * NFFT * 4);
  float* f_fin = (float*)AL((size_t)BB * TT * CC * 4);
  float* f_lam = (float*)AL(256);
  bf16* h_trT = (bf16*)AL((size_t)BB * SQ2 * 2);
  bf16* h_seT = (bf16*)AL((size_t)BB * SQ2 * 2);
  bf16* h_aT  = (bf16*)AL((size_t)BB * SQ2 * 2);
  bf16* h_hA  = (bf16*)AL((size_t)BB * SQ2 * 2);
  bf16* h_dm  = (bf16*)AL((size_t)BB * SQ2 * 2);
  bf16* h_frm = (bf16*)AL((size_t)NFRP * NFFT * 2);
  bf16* h_rim = (bf16*)AL((size_t)NFRP * CP * 2);
  bf16* h_b1  = (bf16*)AL((size_t)CP * NFFT * 2);
  bf16* h_b2  = (bf16*)AL((size_t)NFFT * CP * 2);
  bf16* h_q1w = (bf16*)AL(SQ2 * 2);
  bf16* h_q2w = (bf16*)AL(SQ2 * 2);
  bf16* h_k1w = (bf16*)AL(SQ2 * 2);
  bf16* h_k2w = (bf16*)AL(SQ2 * 2);
  bf16* h_vw  = (bf16*)AL(SQ2 * 2);
  bf16* h_w1  = (bf16*)AL((size_t)FP * TP * 2); // (4T|4C, Kp) weights, reused
  bf16* h_w2  = (bf16*)AL((size_t)TP * FP * 2); // (T|C, 4Tp|4Cp) weights, reused
  // zero-initialized region (pads of GEMM-produced bf16 operands)
  size_t z0 = (off + 255) & ~(size_t)255;
  bf16* h_q1b  = (bf16*)AL((size_t)BB * SQ2 * 2);
  bf16* h_q2b  = (bf16*)AL((size_t)BB * SQ2 * 2);
  bf16* h_k1b  = (bf16*)AL((size_t)BB * SQ2 * 2);
  bf16* h_k2b  = (bf16*)AL((size_t)BB * SQ2 * 2);
  bf16* h_vT   = (bf16*)AL((size_t)BB * SQ2 * 2);
  bf16* h_g2A  = (bf16*)AL((size_t)BB * SQ2 * 2);
  bf16* h_mid  = (bf16*)AL((size_t)BB * CP * FP * 2);
  bf16* h_mid2 = (bf16*)AL((size_t)BB * CP * FP * 2);
  size_t z1 = off;
  hipMemsetAsync((char*)d_ws + z0, 0, z1 - z0, stream);

  // ---- STFT: frames -> (|Z|, angle) via windowed-DFT GEMM ----
  build_basis_fwd_kernel<<<(unsigned)(((size_t)CP * NFFT + 255) / 256), 256, 0, stream>>>(h_b1);
  build_basis_inv_kernel<<<(unsigned)(((size_t)NFFT * CP + 255) / 256), 256, 0, stream>>>(h_b2);
  build_frames_kernel<<<(unsigned)(((size_t)NFRP * NFFT + 255) / 256), 256, 0, stream>>>(IN_x, h_frm);
  launch_gemm(stream, h_frm, 0, NFFT, h_b1, 0, NFFT, f_z, 0, CC,
              nullptr, NFR, CC, NFFT, 1.f, 0, 1);
  magphase_kernel<<<(unsigned)(((size_t)NFR * FRQ + 255) / 256), 256, 0, stream>>>(f_z, f_x);

  const float attn_scale = powf((float)TT, -0.25f);
  const long sB = (long)SQ2;

  for (int l = 0; l < LBLK; ++l) {
    const float* ln_at  = W_lnat + (size_t)l * TT;
    const float* q12w   = W_q12w + (size_t)l * 2 * TT * TT;
    const float* q12b   = W_q12b + (size_t)l * 2 * TT;
    const float* k12w   = W_k12w + (size_t)l * 2 * TT * TT;
    const float* k12b   = W_k12b + (size_t)l * 2 * TT;
    const float* vw_l   = W_vw   + (size_t)l * TT * TT;
    const float* vb_l   = W_vb   + (size_t)l * TT;
    const float* nw_l   = W_nw   + (size_t)l * TT;
    const float* lnff_l = W_lnff + (size_t)l * TT;
    const float* ff1w_l = W_ff1w + (size_t)l * F4T * TT;
    const float* ff1b_l = W_ff1b + (size_t)l * F4T;
    const float* ff2w_l = W_ff2w + (size_t)l * TT * F4T;
    const float* ff2b_l = W_ff2b + (size_t)l * TT;
    const float  lam_init = 0.8f - 0.6f * expf(-0.3f * (float)(l + 1));

    rms_axis1_kernel<<<dim3(CC, BB), 128, 0, stream>>>(f_x, ln_at, f_a);
    decompose_pack_kernel<<<dim3((unsigned)((SQ2 + 255) / 256), BB), 256, 0, stream>>>(f_a, h_trT, h_seT, h_aT);

    launch_packw(stream, q12w,                   TT, TT, TT,  h_q1w, TP, TP);
    launch_packw(stream, q12w + (size_t)TT * TT, TT, TT, TT,  h_q2w, TP, TP);
    launch_packw(stream, k12w,                   TT, TT, TT,  h_k1w, TP, TP);
    launch_packw(stream, k12w + (size_t)TT * TT, TT, TT, TT,  h_k2w, TP, TP);
    launch_packw(stream, vw_l,   TT,  TT,  TT,  h_vw, TP, TP);
    launch_packw(stream, ff1w_l, F4T, TT,  TT,  h_w1, FP, TP);
    launch_packw(stream, ff2w_l, TT,  F4T, F4T, h_w2, TP, FP);

    // projections (per-batch M=C over K=T)
    launch_gemm(stream, h_trT, sB, TP, h_q1w, 0, TP, h_q1b, sB, TP, q12b,      CC, TT, TP, 1.f, EPI_BF16, BB);
    launch_gemm(stream, h_trT, sB, TP, h_q2w, 0, TP, h_q2b, sB, TP, q12b + TT, CC, TT, TP, 1.f, EPI_BF16, BB);
    launch_gemm(stream, h_seT, sB, TP, h_k1w, 0, TP, h_k1b, sB, TP, k12b,      CC, TT, TP, 1.f, EPI_BF16, BB);
    launch_gemm(stream, h_seT, sB, TP, h_k2w, 0, TP, h_k2b, sB, TP, k12b + TT, CC, TT, TP, 1.f, EPI_BF16, BB);
    launch_gemm(stream, h_aT,  sB, TP, h_vw,  0, TP, h_vT,  sB, CP, vb_l,      CC, TT, TP, 1.f, EPI_BF16 | EPI_TRANS, BB);

    lam_kernel<<<1, 512, 0, stream>>>(W_lq1 + (size_t)l * TT, W_lk1 + (size_t)l * TT,
                                      W_lq2 + (size_t)l * TT, W_lk2 + (size_t)l * TT,
                                      lam_init, f_lam);

    // attention scores (C x C over K=T), scaled by T^-0.25
    launch_gemm(stream, h_q1b, sB, TP, h_k1b, sB, TP, f_a1, (long)CC * CC, CC, nullptr, CC, CC, TP, attn_scale, 0, BB);
    launch_gemm(stream, h_q2b, sB, TP, h_k2b, sB, TP, f_a2, (long)CC * CC, CC, nullptr, CC, CC, TP, attn_scale, 0, BB);
    softmax_combine_kernel<<<dim3(CP, BB), 256, 0, stream>>>(f_a1, f_a2, f_lam, h_dm);
    // attn = d @ v  (C x T over K=C)
    launch_gemm(stream, h_dm, sB, CP, h_vT, sB, CP, f_at, (long)CC * TT, TT, nullptr, CC, TT, CP, 1.f, 0, BB);
    rms_add_tr_kernel<<<dim3(CC, BB), 128, 0, stream>>>(f_at, nw_l, 1.f - lam_init, f_x);

    // FFN: rms over t -> 4T silu -> T, accumulated transposed into x
    rms_t_pack_kernel<<<dim3(CP, BB), 128, 0, stream>>>(f_x, lnff_l, h_hA);
    launch_gemm(stream, h_hA, sB, TP, h_w1, 0, TP, h_mid, (long)CP * FP, FP, ff1b_l, CC, F4T, TP, 1.f, EPI_BF16 | EPI_SILU, BB);
    launch_gemm(stream, h_mid, (long)CP * FP, FP, h_w2, 0, FP, f_x, (long)TT * CC, CC, ff2b_l, CC, TT, FP, 1.f, EPI_ACC | EPI_TRANS, BB);
  }

  // ---- head: LN over t, MLP(T), transpose+silu, MLP(C) ----
  ln_t_pack_kernel<<<dim3(CP, BB), 128, 0, stream>>>(f_x, W_lng, W_lnb, h_hA);
  launch_packw(stream, W_llw1, F4T, TT, TT,   h_w1, FP, TP);
  launch_packw(stream, W_llw2, TT, F4T, F4T,  h_w2, TP, FP);
  launch_gemm(stream, h_hA, sB, TP, h_w1, 0, TP, h_mid, (long)CP * FP, FP, W_llb1, CC, F4T, TP, 1.f, EPI_BF16 | EPI_SILU, BB);
  launch_gemm(stream, h_mid, (long)CP * FP, FP, h_w2, 0, FP, h_g2A, sB, CP, W_llb2, CC, TT, FP, 1.f, EPI_BF16 | EPI_TRANS | EPI_SILU, BB);
  launch_packw(stream, W_l2w1, F4C, CC, CC,   h_w1, FP, CP);
  launch_packw(stream, W_l2w2, CC, F4C, F4C,  h_w2, CP, FP);
  launch_gemm(stream, h_g2A, sB, CP, h_w1, 0, CP, h_mid2, (long)CP * FP, FP, W_l2b1, TT, F4C, CP, 1.f, EPI_BF16 | EPI_SILU, BB);
  launch_gemm(stream, h_mid2, (long)CP * FP, FP, h_w2, 0, FP, f_fin, (long)TT * CC, CC, W_l2b2, TT, CC, FP, 1.f, 0, BB);

  // ---- iSTFT: (re|im) GEMM against windowed inverse basis, then overlap-add ----
  polar_pack_kernel<<<(unsigned)(((size_t)NFRP * CP + 255) / 256), 256, 0, stream>>>(f_fin, h_rim);
  launch_gemm(stream, h_rim, 0, CP, h_b2, 0, CP, f_ofr, 0, NFFT, nullptr, NFR, NFFT, CP, 1.f, 0, 1);
  overlap_add_kernel<<<(unsigned)(((size_t)BB * SEQN + 255) / 256), 256, 0, stream>>>(f_ofr, (float*)d_out);
}
